// QNet_91233695302084
// MI455X (gfx1250) — matrix operand order repaired and checked by reading, not compiled, not run on hardware
//
#include <hip/hip_runtime.h>

#define EMB 64
#define NODE_DIM 5
#define TSTEPS 5
#define BB 16
#define NN 512

typedef __attribute__((ext_vector_type(16))) _Float16 v16h;
typedef __attribute__((ext_vector_type(8)))  _Float16 v8h;
typedef __attribute__((ext_vector_type(4)))  _Float16 v4h;
typedef __attribute__((ext_vector_type(8)))  float    v8f;
typedef __attribute__((ext_vector_type(4)))  float    v4f;

// ---------------- P1: s1 = relu(xv@W1a + b1a) @ W1b + b1b ----------------
__global__ __launch_bounds__(EMB) void k_s1(const float* __restrict__ xv,
        const float* __restrict__ W1a, const float* __restrict__ b1a,
        const float* __restrict__ W1b, const float* __restrict__ b1b,
        float* __restrict__ s1) {
    __shared__ float h[EMB];
    const int e = threadIdx.x;
    const int node = blockIdx.x;                 // b*N + n
    const float* x = xv + (size_t)node * NODE_DIM;
    float acc = b1a[e];
    #pragma unroll
    for (int k = 0; k < NODE_DIM; ++k) acc += x[k] * W1a[k * EMB + e];
    h[e] = acc > 0.f ? acc : 0.f;
    __syncthreads();
    float acc2 = b1b[e];
    #pragma unroll 8
    for (int k = 0; k < EMB; ++k) acc2 += h[k] * W1b[k * EMB + e];
    s1[(size_t)node * EMB + e] = acc2;
}

// ------- P2: s3sum[b,j,e] = sum_i relu(Ws[b,i,j]*W4[e] + b4[e]) ----------
__global__ __launch_bounds__(EMB) void k_s3sum(const float* __restrict__ Ws,
        const float* __restrict__ W4, const float* __restrict__ b4,
        float* __restrict__ s3sum) {
    __shared__ float col[NN];
    const int e = threadIdx.x;
    const int b = blockIdx.x >> 9;
    const int j = blockIdx.x & (NN - 1);
    for (int i = e; i < NN; i += EMB)
        col[i] = Ws[((size_t)b * NN + i) * NN + j];   // column j (L2-resident)
    __syncthreads();
    const float w4 = W4[e], bb = b4[e];
    float acc = 0.f;
    #pragma unroll 8
    for (int i = 0; i < NN; ++i) {
        float v = col[i] * w4 + bb;
        acc += v > 0.f ? v : 0.f;
    }
    s3sum[(size_t)blockIdx.x * EMB + e] = acc;
}

// --------- P3: base = s1 + (s3sum @ W3 + b3) + b2  (b2 folded in) --------
__global__ __launch_bounds__(EMB) void k_base(const float* __restrict__ s1,
        const float* __restrict__ s3sum, const float* __restrict__ W3,
        const float* __restrict__ b3, const float* __restrict__ b2,
        float* __restrict__ base) {
    __shared__ float row[EMB];
    const int e = threadIdx.x;
    const int node = blockIdx.x;
    row[e] = s3sum[(size_t)node * EMB + e];
    __syncthreads();
    float acc = b3[e] + b2[e] + s1[(size_t)node * EMB + e];
    #pragma unroll 8
    for (int k = 0; k < EMB; ++k) acc += row[k] * W3[k * EMB + e];
    base[(size_t)node * EMB + e] = acc;
}

// ---------- P0 (optional): connh = (Ws>0) ? 1h : 0h, 4 elems/thread ------
__global__ __launch_bounds__(256) void k_conn(const float* __restrict__ Ws,
        _Float16* __restrict__ connh) {
    const int i = blockIdx.x * 256 + threadIdx.x;
    v4f w = *(const v4f*)(Ws + (size_t)i * 4);
    v4h c;
    #pragma unroll
    for (int q = 0; q < 4; ++q) c[q] = w[q] > 0.f ? (_Float16)1.f : (_Float16)0.f;
    *(v4h*)(connh + (size_t)i * 4) = c;
}

// ------------------ t=0 step: mu = relu(base) (mu0 == 0) -----------------
__global__ __launch_bounds__(256) void k_init_mu(const float* __restrict__ base,
        float* __restrict__ mu, int n) {
    int i = blockIdx.x * 256 + threadIdx.x;
    if (i < n) { float v = base[i]; mu[i] = v > 0.f ? v : 0.f; }
}

// -------- A-step: zhT[b,e,j] = f16( sum_k mu[b,j,k] * W2[k,e] ) ----------
__global__ __launch_bounds__(256) void k_z(const float* __restrict__ mu,
        const float* __restrict__ W2, _Float16* __restrict__ zhT) {
    const int idx = blockIdx.x * 256 + threadIdx.x;  // b*E*N + e*N + j
    const int j = idx & (NN - 1);
    const int e = (idx >> 9) & (EMB - 1);
    const int b = idx >> 15;
    const float* m = mu + ((size_t)b * NN + j) * EMB;
    float acc = 0.f;
    #pragma unroll 8
    for (int k = 0; k < EMB; ++k) acc += m[k] * W2[k * EMB + e];
    zhT[idx] = (_Float16)acc;
}

// ---- B-step (WMMA): mu = relu(base + conn @ zh) -------------------------
// PRE=true : A-tiles from precomputed f16 connh (2x b128 loads/lane/kstep)
// PRE=false: A-tiles built on the fly from f32 Ws (cmp+cndmask)
// 4 waves/block, each wave computes a 16x64 tile of one batch.
template <bool PRE>
__global__ __launch_bounds__(128) void k_iter_wmma(const float* __restrict__ Ws,
        const _Float16* __restrict__ connh,
        const _Float16* __restrict__ zhT, const float* __restrict__ base,
        float* __restrict__ mu) {
    const int lane = threadIdx.x & 31;
    const int wave = threadIdx.x >> 5;
    const int b    = blockIdx.x;
    const int i0   = blockIdx.y * 64 + wave * 16;
    const int row  = lane & 15;
    const int half = lane >> 4;

    const float*    Arow  = Ws    + ((size_t)b * NN + (i0 + row)) * NN;
    const _Float16* Hrow  = connh + ((size_t)b * NN + (i0 + row)) * NN;
    const _Float16* Bbase = zhT   + (size_t)b * EMB * NN;

    v8f acc[4] = {};
    #pragma unroll 2
    for (int j0 = 0; j0 < NN; j0 += 32) {
        // A tile: ISA 16-bit A layout = two contiguous 8-element chunks/lane
        v16h a;
        if (PRE) {
            v8h alo = *(const v8h*)(Hrow + j0 + half * 8);
            v8h ahi = *(const v8h*)(Hrow + j0 + 16 + half * 8);
            a = __builtin_shufflevector(alo, ahi,
                    0, 1, 2, 3, 4, 5, 6, 7, 8, 9, 10, 11, 12, 13, 14, 15);
        } else {
            v8f wlo = *(const v8f*)(Arow + j0 + half * 8);
            v8f whi = *(const v8f*)(Arow + j0 + 16 + half * 8);
            #pragma unroll
            for (int q = 0; q < 8; ++q) {
                a[q]     = wlo[q] > 0.f ? (_Float16)1.f : (_Float16)0.f;
                a[8 + q] = whi[q] > 0.f ? (_Float16)1.f : (_Float16)0.f;
            }
        }
        // B tiles: zhT rows give two 8-element K chunks per lane (same layout as A)
        #pragma unroll
        for (int et = 0; et < 4; ++et) {
            const _Float16* bp = Bbase + (size_t)(et * 16 + row) * NN + j0 + half * 8;
            v8h blo = *(const v8h*)bp;
            v8h bhi = *(const v8h*)(bp + 16);
            v16h bm = __builtin_shufflevector(blo, bhi,
                        0, 1, 2, 3, 4, 5, 6, 7, 8, 9, 10, 11, 12, 13, 14, 15);
            acc[et] = __builtin_amdgcn_wmma_f32_16x16x32_f16(
                          false, a, false, bm, (short)0, acc[et], false, false);
        }
    }
    // D layout: VGPR r -> row r + 8*half, col = et*16 + row
    #pragma unroll
    for (int et = 0; et < 4; ++et) {
        #pragma unroll
        for (int r = 0; r < 8; ++r) {
            size_t o = ((size_t)b * NN + i0 + half * 8 + r) * EMB + et * 16 + row;
            float v = acc[et][r] + base[o];
            mu[o] = v > 0.f ? v : 0.f;
        }
    }
}

// ---- F1: gscal[b] = b5 + sum_e relu((sum_n mu)@W6 + b6)[e] * W5[e] ------
__global__ __launch_bounds__(EMB) void k_glob(const float* __restrict__ mu,
        const float* __restrict__ W6, const float* __restrict__ b6,
        const float* __restrict__ W5, const float* __restrict__ b5,
        float* __restrict__ gscal) {
    __shared__ float ms[EMB];
    __shared__ float red[EMB];
    const int e = threadIdx.x, b = blockIdx.x;
    const float* m = mu + (size_t)b * NN * EMB;
    float s = 0.f;
    for (int n = 0; n < NN; ++n) s += m[n * EMB + e];
    ms[e] = s;
    __syncthreads();
    float g = b6[e];
    #pragma unroll 8
    for (int k = 0; k < EMB; ++k) g += ms[k] * W6[k * EMB + e];
    g = g > 0.f ? g : 0.f;
    red[e] = g * W5[e];
    __syncthreads();
    for (int s2 = EMB / 2; s2 > 0; s2 >>= 1) {
        if (e < s2) red[e] += red[e + s2];
        __syncthreads();
    }
    if (e == 0) gscal[b] = red[0] + b5[0];
}

// ---- F2: out[b,n] = gscal[b] + sum_e relu(mu@W7 + b7)[e] * W5[64+e] -----
__global__ __launch_bounds__(EMB) void k_out(const float* __restrict__ mu,
        const float* __restrict__ W7, const float* __restrict__ b7,
        const float* __restrict__ W5, const float* __restrict__ gscal,
        float* __restrict__ out) {
    __shared__ float m[EMB];
    __shared__ float red[EMB];
    const int e = threadIdx.x;
    const int node = blockIdx.x;
    const int b = node >> 9;
    m[e] = mu[(size_t)node * EMB + e];
    __syncthreads();
    float l = b7[e];
    #pragma unroll 8
    for (int k = 0; k < EMB; ++k) l += m[k] * W7[k * EMB + e];
    l = l > 0.f ? l : 0.f;
    red[e] = l * W5[EMB + e];
    __syncthreads();
    for (int s2 = EMB / 2; s2 > 0; s2 >>= 1) {
        if (e < s2) red[e] += red[e + s2];
        __syncthreads();
    }
    if (e == 0) out[node] = gscal[b] + red[0];
}

extern "C" void kernel_launch(void* const* d_in, const int* in_sizes, int n_in,
                              void* d_out, int out_size, void* d_ws, size_t ws_size,
                              hipStream_t stream) {
    const float* xv  = (const float*)d_in[0];
    const float* Ws  = (const float*)d_in[1];
    const float* W1a = (const float*)d_in[2];
    const float* b1a = (const float*)d_in[3];
    const float* W1b = (const float*)d_in[4];
    const float* b1b = (const float*)d_in[5];
    const float* W2  = (const float*)d_in[6];
    const float* b2  = (const float*)d_in[7];
    const float* W3  = (const float*)d_in[8];
    const float* b3  = (const float*)d_in[9];
    const float* W4  = (const float*)d_in[10];
    const float* b4  = (const float*)d_in[11];
    const float* W5  = (const float*)d_in[12];
    const float* b5  = (const float*)d_in[13];
    const float* W6  = (const float*)d_in[14];
    const float* b6  = (const float*)d_in[15];
    const float* W7  = (const float*)d_in[16];
    const float* b7  = (const float*)d_in[17];
    float* out = (float*)d_out;

    const size_t SZ = (size_t)BB * NN * EMB * sizeof(float);   // 2 MB
    char* w = (char*)d_ws;
    float*    s1    = (float*)(w);
    float*    s3sum = (float*)(w + SZ);
    float*    base  = (float*)(w + 2 * SZ);
    float*    mu    = (float*)(w + 3 * SZ);
    _Float16* zhT   = (_Float16*)(w + 4 * SZ);                 // 1 MB
    float*    gscal = (float*)(w + 4 * SZ + SZ / 2);
    const size_t connOff  = 4 * SZ + SZ / 2 + 256;             // 256B-aligned
    _Float16* connh = (_Float16*)(w + connOff);
    const size_t connBytes = (size_t)BB * NN * NN * sizeof(_Float16);  // 8 MB
    const bool pre = ws_size >= connOff + connBytes;           // fixed per run
    (void)in_sizes; (void)n_in; (void)out_size;

    k_s1   <<<BB * NN, EMB, 0, stream>>>(xv, W1a, b1a, W1b, b1b, s1);
    k_s3sum<<<BB * NN, EMB, 0, stream>>>(Ws, W4, b4, s3sum);
    k_base <<<BB * NN, EMB, 0, stream>>>(s1, s3sum, W3, b3, b2, base);
    if (pre)
        k_conn<<<(BB * NN * NN / 4) / 256, 256, 0, stream>>>(Ws, connh);
    k_init_mu<<<(BB * NN * EMB) / 256, 256, 0, stream>>>(base, mu, BB * NN * EMB);
    for (int t = 1; t < TSTEPS; ++t) {
        k_z<<<(BB * EMB * NN) / 256, 256, 0, stream>>>(mu, W2, zhT);
        if (pre)
            k_iter_wmma<true><<<dim3(BB, NN / 64), 128, 0, stream>>>(Ws, connh, zhT, base, mu);
        else
            k_iter_wmma<false><<<dim3(BB, NN / 64), 128, 0, stream>>>(Ws, connh, zhT, base, mu);
    }
    k_glob<<<BB, EMB, 0, stream>>>(mu, W6, b6, W5, b5, gscal);
    k_out <<<BB * NN, EMB, 0, stream>>>(mu, W7, b7, W5, gscal, out);
}
